// PatchModule_12618613916406
// MI455X (gfx1250) — compile-verified, matrix operand order checked by reading, not run on hardware
//
#include <hip/hip_runtime.h>
#include <stdint.h>

// ---------------------------------------------------------------------------
// CDNA5 (gfx1250) implementation: conv3x3+BN+ReLU towers as implicit GEMM on
// v_wmma_f32_16x16x32_bf16 (wave32), plus small epilogue kernels.
// Block = 256 threads (8 wave32) computes 32 pixels x 256 out-channels.
// Each wave: 2 M-tiles x 2 N-tiles = 4 f32 accumulators, 4 WMMA per K-step.
// K loop: 9 taps x 8 chunks of 32 input channels, double-buffered A in LDS,
// one barrier per K-step. B fragments stream from L2 (bf16 weights, 3.4MB)
// in a fragment-native layout so every lane load is part of a contiguous
// 256B burst per half-wave.
// ---------------------------------------------------------------------------

typedef __attribute__((ext_vector_type(16))) __bf16 v16bf;
typedef __attribute__((ext_vector_type(8)))  float  v8f;

union FragBF { uint4 u[2]; v16bf v; };

__device__ __forceinline__ unsigned short f2bf(float f) {
  union { float f; unsigned u; } x; x.f = f;
  unsigned u = x.u;
  return (unsigned short)((u + 0x7FFFu + ((u >> 16) & 1u)) >> 16); // RNE
}

// ---------------------------------------------------------------------------
// Weight prep: OIHW f32 -> bf16 fragment-native layout
//   dst[(((tap*8 + kc)*4 + q)*256 + o)*8 + k8]  with input chan i = kc*32+q*8+k8
// so a WMMA B fragment chunk (8 consecutive K for one output column o) is one
// 16B load, and consecutive lanes (o) read consecutive 16B chunks.
// ---------------------------------------------------------------------------
__global__ __launch_bounds__(256) void prep_w(const float* __restrict__ src,
                                              unsigned short* __restrict__ dst) {
  int e = blockIdx.x * 256 + threadIdx.x;
  if (e >= 9 * 256 * 256) return;
  const int k8  = e & 7;
  const int o   = (e >> 3) & 255;
  const int q   = (e >> 11) & 3;
  const int kc  = (e >> 13) & 7;
  const int tap = e >> 16;
  const int i   = kc * 32 + q * 8 + k8;
  dst[e] = f2bf(src[o * 2304 + i * 9 + tap]);  // src[o][i][ky][kx]
}

// Fold BN (per level) + conv bias into per-channel scale/bias.
__global__ __launch_bounds__(256) void prep_sb(const float* __restrict__ b,
                                               const float* __restrict__ g,
                                               const float* __restrict__ bb,
                                               const float* __restrict__ m,
                                               const float* __restrict__ v,
                                               float* __restrict__ sb) {
  int e = blockIdx.x * 256 + threadIdx.x;      // 3 levels * 256 channels
  if (e >= 768) return;
  int c = e & 255, lvl = e >> 8;
  float s = g[lvl * 256 + c] * rsqrtf(v[lvl * 256 + c] + 1e-5f);
  sb[lvl * 512 + c]       = s;
  sb[lvl * 512 + 256 + c] = (b[c] - m[lvl * 256 + c]) * s + bb[lvl * 256 + c];
}

// ---------------------------------------------------------------------------
// conv3x3 (pad=1) + folded BN + ReLU via WMMA implicit GEMM.
// ---------------------------------------------------------------------------
__global__ __launch_bounds__(256) void conv3x3_bn_relu(
    const float* __restrict__ in, const unsigned short* __restrict__ wqT,
    const float* __restrict__ sb, float* __restrict__ out,
    int H, int W) {
  const int tilesX = W >> 5;
  int t = blockIdx.x;
  const int xt = t % tilesX; t /= tilesX;
  const int y  = t % H;
  const int b  = t / H;
  const int x0 = xt << 5;
  const long HW = (long)H * W;

  // Double-buffered A tile: 32 pixels x 32 K (bf16), row stride 48 keeps
  // the per-lane 16B fragment chunks 16B-aligned.
  __shared__ __align__(16) unsigned short A[2][32][48];

  const int tid  = threadIdx.x;
  const int lane = tid & 31;
  const int wv   = tid >> 5;        // wave 0..7 -> N-tiles 2w, 2w+1
  const int col  = lane & 15;       // M/N index within a 16-wide tile
  const int half = lane >> 4;       // K split selector

  v8f acc00 = {}, acc01 = {}, acc10 = {}, acc11 = {};
  const int n0a = wv * 32;
  const int n0b = wv * 32 + 16;

  const int ai = tid & 31;          // pixel this thread stages
  const int ag = tid >> 5;          // channel group: channels 4*ag .. 4*ag+3
  const float* inB = in + (long)b * 256 * HW;

  // Stage A tile for K-step s into LDS buffer bf. Branch-free boundary
  // handling: clamp coords (always-in-bounds load) and multiply by 0/1 mask.
  auto stageA = [&](int s, int bf) {
    const int tap = s >> 3, kc = s & 7;
    const int dy = tap / 3 - 1, dx = tap % 3 - 1;
    const int yy = y + dy;
    const int xx = x0 + ai + dx;
    const float okf =
        (((unsigned)yy < (unsigned)H) & ((unsigned)xx < (unsigned)W)) ? 1.f : 0.f;
    const int yyc = yy < 0 ? 0 : (yy >= H ? H - 1 : yy);
    const int xxc = xx < 0 ? 0 : (xx >= W ? W - 1 : xx);
    const float* p = inB + (long)(kc * 32 + 4 * ag) * HW + (long)yyc * W + xxc;
    const float f0 = p[0] * okf;
    const float f1 = p[HW] * okf;
    const float f2 = p[2 * HW] * okf;
    const float f3 = p[3 * HW] * okf;
    uint2 st;
    st.x = (unsigned)f2bf(f0) | ((unsigned)f2bf(f1) << 16);
    st.y = (unsigned)f2bf(f2) | ((unsigned)f2bf(f3) << 16);
    *(uint2*)&A[bf][ai][4 * ag] = st;   // one ds_store_b64
  };

  stageA(0, 0);

  for (int s = 0; s < 72; ++s) {
    __syncthreads();                    // single barrier per K-step
    const int bf = s & 1;
    if (s + 1 < 72) stageA(s + 1, bf ^ 1);

    FragBF a0, a1, b0, b1;
    a0.u[0] = *(const uint4*)&A[bf][col][half * 8];
    a0.u[1] = *(const uint4*)&A[bf][col][16 + half * 8];
    a1.u[0] = *(const uint4*)&A[bf][16 + col][half * 8];
    a1.u[1] = *(const uint4*)&A[bf][16 + col][16 + half * 8];
    // Fragment-native weights: chunk q = half (K 0..15 part) and q = 2+half.
    const unsigned short* wlo = wqT + ((long)s * 4 + half) * 2048;
    const unsigned short* whi = wlo + 2 * 2048;
    b0.u[0] = *(const uint4*)(wlo + (n0a + col) * 8);
    b0.u[1] = *(const uint4*)(whi + (n0a + col) * 8);
    b1.u[0] = *(const uint4*)(wlo + (n0b + col) * 8);
    b1.u[1] = *(const uint4*)(whi + (n0b + col) * 8);

    acc00 = __builtin_amdgcn_wmma_f32_16x16x32_bf16(false, a0.v, false, b0.v,
                                                    (short)0, acc00, false, false);
    acc01 = __builtin_amdgcn_wmma_f32_16x16x32_bf16(false, a0.v, false, b1.v,
                                                    (short)0, acc01, false, false);
    acc10 = __builtin_amdgcn_wmma_f32_16x16x32_bf16(false, a1.v, false, b0.v,
                                                    (short)0, acc10, false, false);
    acc11 = __builtin_amdgcn_wmma_f32_16x16x32_bf16(false, a1.v, false, b1.v,
                                                    (short)0, acc11, false, false);
  }

  // Epilogue: D layout => lane holds column co, VGPR r holds pixel m=r+8*half.
  auto storeTile = [&](v8f a, int co, int mt) {
    const float s  = sb[co];
    const float bi = sb[256 + co];
    float* o = out + ((long)b * 256 + co) * HW + (long)y * W + x0 + mt * 16 + half * 8;
    float4 lo, hi;
    float* pl = (float*)&lo;
    float* ph = (float*)&hi;
#pragma unroll
    for (int r = 0; r < 4; ++r) {
      float x = a[r] * s + bi;       pl[r] = x > 0.f ? x : 0.f;
      x = a[r + 4] * s + bi;         ph[r] = x > 0.f ? x : 0.f;
    }
    *(float4*)o       = lo;
    *(float4*)(o + 4) = hi;
  };
  storeTile(acc00, n0a + col, 0);
  storeTile(acc01, n0b + col, 0);
  storeTile(acc10, n0a + col, 1);
  storeTile(acc11, n0b + col, 1);
}

// ---------------------------------------------------------------------------
// 1x1 conv (256 -> 1) + sigmoid. One thread per 4 pixels, float4-coalesced.
// ---------------------------------------------------------------------------
__global__ __launch_bounds__(256) void ehead_sigmoid(
    const float* __restrict__ ex, const float* __restrict__ w,
    const float* __restrict__ b, float* __restrict__ em, int HW) {
  const int quads = HW >> 2;
  long q = (long)blockIdx.x * 256 + threadIdx.x;
  if (q >= 4L * quads) return;
  const int bb = (int)(q / quads);
  const int pix = (int)(q % quads) * 4;
  const float* base = ex + (long)bb * 256 * HW + pix;
  const float b0 = b[0];
  float4 acc = {b0, b0, b0, b0};
  for (int c = 0; c < 256; ++c) {
    const float4 x = *(const float4*)(base + (long)c * HW);
    const float wc = w[c];
    acc.x += x.x * wc; acc.y += x.y * wc;
    acc.z += x.z * wc; acc.w += x.w * wc;
  }
  float* o = em + (long)bb * HW + pix;
  o[0] = 1.f / (1.f + __expf(-acc.x));
  o[1] = 1.f / (1.f + __expf(-acc.y));
  o[2] = 1.f / (1.f + __expf(-acc.z));
  o[3] = 1.f / (1.f + __expf(-acc.w));
}

// Gather entropy at the 1024 candidate anchors per batch.
__global__ __launch_bounds__(256) void gather_ape(
    const float* __restrict__ em, const int* __restrict__ tc,
    float* __restrict__ ape, int H, int W) {
  int i = blockIdx.x * 256 + threadIdx.x;   // 4 * 1024
  if (i >= 4096) return;
  int b = i >> 10, p = i & 1023;
  ape[i] = em[(long)b * H * W + (long)tc[p] * W + tc[1024 + p]];
}

// Exact lax.top_k: rank = #(greater) + #(equal with lower index).
__global__ __launch_bounds__(1024) void topk_rank(
    const float* __restrict__ ape, int* __restrict__ topi, int nk) {
  __shared__ float vals[1024];
  const int b = blockIdx.x, t = threadIdx.x;
  vals[t] = ape[b * 1024 + t];
  __syncthreads();
  const float v = vals[t];
  int rank = 0;
  for (int j = 0; j < 1024; ++j) {
    float u = vals[j];
    rank += (u > v) || (u == v && j < t);
  }
  if (rank < nk) topi[b * nk + rank] = t;
}

// Gather feature, 256->4 head + sigmoid, box composition. Block per (b,k).
__global__ __launch_bounds__(256) void head_kernel(
    const float* __restrict__ px, const int* __restrict__ topi,
    const int* __restrict__ tc, const float* __restrict__ hw,
    const float* __restrict__ hb, float* __restrict__ fin,
    float* __restrict__ xy, int nk, int H, int W) {
  __shared__ float red[256];
  __shared__ float pcs[4];
  const int b = blockIdx.x / nk, k = blockIdx.x % nk, t = threadIdx.x;
  const int p = topi[b * nk + k];
  const int r = tc[p], c = tc[1024 + p];
  const long HW = (long)H * W;
  const float a = px[((long)b * 256 + t) * HW + (long)r * W + c];
  for (int j = 0; j < 4; ++j) {
    red[t] = a * hw[j * 256 + t];
    __syncthreads();
    for (int s = 128; s > 0; s >>= 1) {
      if (t < s) red[t] += red[t + s];
      __syncthreads();
    }
    if (t == 0) pcs[j] = 1.f / (1.f + __expf(-(red[0] + hb[j])));
    __syncthreads();
  }
  if (t == 0) {
    const float ax = (float)r, ay = (float)c;
    float* f = fin + ((long)b * nk + k) * 4;
    f[0] = ax - 16.f * pcs[0];
    f[1] = ay - 16.f * pcs[1];
    f[2] = ax + 16.f * pcs[2];
    f[3] = ay + 16.f * pcs[3];
    float* q = xy + ((long)b * nk + k) * 2;
    q[0] = ax; q[1] = ay;
  }
}

// ---------------------------------------------------------------------------
// Host-side orchestration.
// Workspace layout (bytes):  [xs 64MB][t2 64MB][wq bf16 3.4MB][sb][ape][topi]
// ---------------------------------------------------------------------------
static constexpr size_t OFF_XS   = 0;
static constexpr size_t OFF_T2   = 67108864;
static constexpr size_t OFF_WQ   = 134217728;              // 3 * 589824 ushorts
static constexpr size_t OFF_SB   = OFF_WQ + 3u * 1179648;  // 3 * 1536 floats
static constexpr size_t OFF_APE  = OFF_SB + 18432;
static constexpr size_t OFF_TOPI = OFF_APE + 16384;

extern "C" void kernel_launch(void* const* d_in, const int* in_sizes, int n_in,
                              void* d_out, int out_size, void* d_ws, size_t ws_size,
                              hipStream_t stream) {
  (void)in_sizes; (void)n_in; (void)out_size; (void)ws_size;
  const float* feat[3] = {(const float*)d_in[0], (const float*)d_in[1],
                          (const float*)d_in[2]};
  const int* tc = (const int*)d_in[3];
  // tower params: share(4..9), ent(10..15), pat(16..21) -> w,b,g,bb,m,v
  const float *tw[3], *tb[3], *tg[3], *tbb[3], *tm[3], *tv[3];
  for (int tt = 0; tt < 3; ++tt) {
    int base = 4 + 6 * tt;
    tw[tt]  = (const float*)d_in[base + 0];
    tb[tt]  = (const float*)d_in[base + 1];
    tg[tt]  = (const float*)d_in[base + 2];
    tbb[tt] = (const float*)d_in[base + 3];
    tm[tt]  = (const float*)d_in[base + 4];
    tv[tt]  = (const float*)d_in[base + 5];
  }
  const float* ehw = (const float*)d_in[22];
  const float* ehb = (const float*)d_in[23];
  const float* hw  = (const float*)d_in[24];
  const float* hb  = (const float*)d_in[25];
  float* out = (float*)d_out;
  char*  ws  = (char*)d_ws;

  float*          xs   = (float*)(ws + OFF_XS);
  float*          t2   = (float*)(ws + OFF_T2);
  unsigned short* wq   = (unsigned short*)(ws + OFF_WQ);
  float*          sb   = (float*)(ws + OFF_SB);
  float*          ape  = (float*)(ws + OFF_APE);
  int*            topi = (int*)(ws + OFF_TOPI);

  for (int tt = 0; tt < 3; ++tt) {
    prep_w<<<2304, 256, 0, stream>>>(tw[tt], wq + (size_t)tt * 589824);
    prep_sb<<<3, 256, 0, stream>>>(tb[tt], tg[tt], tbb[tt], tm[tt], tv[tt],
                                   sb + tt * 1536);
  }

  const int NKs[3]    = {512, 256, 128};
  const int emoff[3]  = {0, 77824, 100352};
  const int finoff[3] = {65536, 94208, 104448};
  const int xyoff[3]  = {73728, 98304, 106496};

  for (int lvl = 0; lvl < 3; ++lvl) {
    const int H = 128 >> lvl, W = H, nk = NKs[lvl];
    const int HW = H * W;
    const int grid = 4 * H * (W / 32);
    // xs = share tower
    conv3x3_bn_relu<<<grid, 256, 0, stream>>>(feat[lvl], wq,
                                              sb + 0 * 1536 + lvl * 512, xs, H, W);
    // ex = ent tower
    conv3x3_bn_relu<<<grid, 256, 0, stream>>>(xs, wq + 589824,
                                              sb + 1 * 1536 + lvl * 512, t2, H, W);
    ehead_sigmoid<<<(4 * (HW / 4) + 255) / 256, 256, 0, stream>>>(
        t2, ehw, ehb, out + emoff[lvl], HW);
    gather_ape<<<16, 256, 0, stream>>>(out + emoff[lvl], tc, ape, H, W);
    topk_rank<<<4, 1024, 0, stream>>>(ape, topi, nk);
    // px = pat tower (reuses t2 after em/ape extracted)
    conv3x3_bn_relu<<<grid, 256, 0, stream>>>(xs, wq + 2 * 589824,
                                              sb + 2 * 1536 + lvl * 512, t2, H, W);
    head_kernel<<<4 * nk, 256, 0, stream>>>(t2, topi, tc, hw, hb,
                                            out + finoff[lvl], out + xyoff[lvl],
                                            nk, H, W);
  }
}
